// GATv2Net_34857954574553
// MI455X (gfx1250) — matrix operand order repaired and checked by reading, not compiled, not run on hardware
//
#include <hip/hip_runtime.h>
#include <hip/hip_bf16.h>
#include <math.h>

#define NNODES 50000
#define NEDGES 800000
#define CH 128
#define NEGS 0.2f

typedef __attribute__((ext_vector_type(16))) __bf16 v16bf;
typedef __attribute__((ext_vector_type(8)))  float  v8f;

// ---------------- WMMA GEMM: out[r,n] = sum_k in[r,k] * wt[n,k]  (wt = W^T, 128x128)
__global__ __launch_bounds__(256) void gemm128_bf16(const float* __restrict__ in,
                                                    const float* __restrict__ wt,
                                                    float* __restrict__ out,
                                                    int nrows) {
  const int wave  = threadIdx.x >> 5;
  const int mtile = blockIdx.x * 8 + wave;
  if (mtile * 16 >= nrows) return;               // wave-uniform: EXEC stays all-1s for WMMA
  const int lane = threadIdx.x & 31;
  const int half = lane >> 4;
  const int mr   = lane & 15;

  const float* arow = in + (size_t)(mtile * 16 + mr) * CH;

  // A fragments: 16-bit A 16x32 layout. lanes0-15: M=lane, K pairs starting at 8*half and 16+8*half
  v16bf afrag[4];
#pragma unroll
  for (int ks = 0; ks < 4; ++ks) {
    const float* p = arow + ks * 32 + half * 8;
    const float* q = arow + ks * 32 + 16 + half * 8;
#pragma unroll
    for (int j = 0; j < 8; ++j) {
      afrag[ks][j]     = (__bf16)p[j];
      afrag[ks][j + 8] = (__bf16)q[j];
    }
  }

#pragma unroll
  for (int nt = 0; nt < 8; ++nt) {
    v8f acc = {};
    const float* wr = wt + (size_t)(nt * 16 + mr) * CH;   // column (nt*16+mr) of W, K-contiguous
#pragma unroll
    for (int ks = 0; ks < 4; ++ks) {
      // B 32x16 layout: lane holds column mr; K = ks*32 + 8*half + j (j<8), ks*32 + 16 + 8*half + (j-8) (j>=8)
      v16bf b;
      const float* p = wr + ks * 32 + half * 8;
#pragma unroll
      for (int j = 0; j < 16; ++j) b[j] = (__bf16)p[j < 8 ? j : j + 8];
      acc = __builtin_amdgcn_wmma_f32_16x16x32_bf16(false, afrag[ks], false, b,
                                                    (short)0, acc, false, false);
    }
    // D: VGPR v -> row (v + 8*half), col = mr
    float* obase = out + (size_t)(mtile * 16) * CH + nt * 16;
#pragma unroll
    for (int v = 0; v < 8; ++v) obase[(size_t)(v + 8 * half) * CH + mr] = acc[v];
  }
}

// ---------------- helpers
__global__ void transpose128(const float* __restrict__ w, float* __restrict__ wt) {
  int t = blockIdx.x * blockDim.x + threadIdx.x;
  if (t >= CH * CH) return;
  int k = t >> 7, n = t & 127;
  wt[n * CH + k] = w[k * CH + n];
}

__global__ void fillf(float* __restrict__ p, float v, int n) {
  int t = blockIdx.x * blockDim.x + threadIdx.x;
  if (t < n) p[t] = v;
}

__device__ __forceinline__ void atomicMaxF(float* a, float v) {
  if (v >= 0.0f) atomicMax((int*)a, __float_as_int(v));
  else           atomicMin((unsigned int*)a, (unsigned int)__float_as_int(v));
}

// ---------------- layer-1 edge scores: 128 threads/edge; each wave = one head (32 ch)
__global__ void edge_score_h4(const float* __restrict__ xl, const float* __restrict__ xr,
                              const int* __restrict__ src, const int* __restrict__ dst,
                              const float* __restrict__ att, float* __restrict__ esc,
                              float* __restrict__ mx, int E) {
  int t = blockIdx.x * blockDim.x + threadIdx.x;
  int e = t >> 7;
  if (e >= E) return;
  int c = t & 127;
  int h = c >> 5;
  int s = src[e], d = dst[e];
  float m = xl[(size_t)s * CH + c] + xr[(size_t)d * CH + c];
  m = (m > 0.0f) ? m : NEGS * m;
  float v = m * att[c];
#pragma unroll
  for (int off = 16; off > 0; off >>= 1) v += __shfl_down(v, off, 32);
  if ((threadIdx.x & 31) == 0) {
    esc[(size_t)e * 4 + h] = v;
    atomicMaxF(&mx[(size_t)d * 4 + h], v);
  }
}

// ---------------- layer-2 edge scores: 1 wave/edge, 4 channels/lane
__global__ void edge_score_h1(const float* __restrict__ hl, const float* __restrict__ hr,
                              const int* __restrict__ src, const int* __restrict__ dst,
                              const float* __restrict__ att, float* __restrict__ esc,
                              float* __restrict__ mx, int E) {
  int wid = (blockIdx.x * blockDim.x + threadIdx.x) >> 5;
  if (wid >= E) return;
  int lane = threadIdx.x & 31;
  int s = src[wid], d = dst[wid];
  const float4 a = ((const float4*)(hl + (size_t)s * CH))[lane];
  const float4 b = ((const float4*)(hr + (size_t)d * CH))[lane];
  const float4 w = ((const float4*)att)[lane];
  float v = 0.0f, m;
  m = a.x + b.x; m = (m > 0.f) ? m : NEGS * m; v += m * w.x;
  m = a.y + b.y; m = (m > 0.f) ? m : NEGS * m; v += m * w.y;
  m = a.z + b.z; m = (m > 0.f) ? m : NEGS * m; v += m * w.z;
  m = a.w + b.w; m = (m > 0.f) ? m : NEGS * m; v += m * w.w;
#pragma unroll
  for (int off = 16; off > 0; off >>= 1) v += __shfl_down(v, off, 32);
  if (lane == 0) {
    esc[wid] = v;
    atomicMaxF(&mx[d], v);
  }
}

// ---------------- softmax numerator + segment sum (H = 1<<hshift entries/node)
__global__ void edge_exp(float* __restrict__ esc, const float* __restrict__ mx,
                         float* __restrict__ denom, const int* __restrict__ dst,
                         int E, int hshift) {
  int t = blockIdx.x * blockDim.x + threadIdx.x;
  if (t >= (E << hshift)) return;
  int e = t >> hshift;
  int h = t & ((1 << hshift) - 1);
  int d = dst[e];
  float ex = __expf(esc[t] - mx[((size_t)d << hshift) + h]);
  esc[t] = ex;
  atomicAdd(&denom[((size_t)d << hshift) + h], ex);
}

// ---------------- weighted scatter aggregate: 128 threads/edge
__global__ void edge_agg(const float* __restrict__ xl, const float* __restrict__ esc,
                         const float* __restrict__ denom, const int* __restrict__ src,
                         const int* __restrict__ dst, float* __restrict__ out,
                         int E, int hshift, int cshift) {
  int t = blockIdx.x * blockDim.x + threadIdx.x;
  int e = t >> 7;
  if (e >= E) return;
  int c = t & 127;
  int h = c >> cshift;              // cshift=5 -> head=c/32 ; cshift=7 -> head=0
  int s = src[e], d = dst[e];
  float den = denom[((size_t)d << hshift) + h];
  float alpha = esc[((size_t)e << hshift) + h] / ((den == 0.0f) ? 1.0f : den);
  atomicAdd(&out[(size_t)d * CH + c], alpha * xl[(size_t)s * CH + c]);
}

__global__ void elu_bias(float* __restrict__ p, const float* __restrict__ b, int n) {
  int t = blockIdx.x * blockDim.x + threadIdx.x;
  if (t >= n) return;
  float v = p[t] + b[t & 127];
  p[t] = (v > 0.0f) ? v : (__expf(v) - 1.0f);
}

__global__ void bias_add(float* __restrict__ p, const float* __restrict__ b, int n) {
  int t = blockIdx.x * blockDim.x + threadIdx.x;
  if (t >= n) return;
  p[t] += b[t & 127];
}

extern "C" void kernel_launch(void* const* d_in, const int* in_sizes, int n_in,
                              void* d_out, int out_size, void* d_ws, size_t ws_size,
                              hipStream_t stream) {
  const float* x    = (const float*)d_in[0];
  const int*   ei   = (const int*)  d_in[1];
  const float* W1l  = (const float*)d_in[2];
  const float* W1r  = (const float*)d_in[3];
  const float* att1 = (const float*)d_in[4];
  const float* b1   = (const float*)d_in[5];
  const float* W2l  = (const float*)d_in[6];
  const float* W2r  = (const float*)d_in[7];
  const float* att2 = (const float*)d_in[8];
  const float* b2   = (const float*)d_in[9];
  float* out = (float*)d_out;

  const int N = NNODES, E = NEDGES;
  const int* src = ei;
  const int* dst = ei + E;

  float* ws    = (float*)d_ws;
  float* xl    = ws;                          // N*128   (layer1 xl, layer2 hl)
  float* xr    = xl    + (size_t)N * CH;      // N*128   (layer1 xr, layer2 hr)
  float* hbuf  = xr    + (size_t)N * CH;      // N*128   (agg1 -> elu -> h)
  float* esc   = hbuf  + (size_t)N * CH;      // E*4
  float* mx    = esc   + (size_t)E * 4;       // N*4
  float* denom = mx    + (size_t)N * 4;       // N*4
  float* wt    = denom + (size_t)N * 4;       // 128*128 (W^T scratch)

  const int TB = 256;
  dim3 blk(TB);
  const int gemm_blocks = ((N / 16) + 7) / 8;     // 3125 M-tiles, 8 waves/block
  const int tw_blocks   = (CH * CH + TB - 1) / TB;

  // ---------- layer 1: GEMMs ----------
  transpose128<<<tw_blocks, blk, 0, stream>>>(W1l, wt);
  gemm128_bf16<<<gemm_blocks, blk, 0, stream>>>(x, wt, xl, N);
  transpose128<<<tw_blocks, blk, 0, stream>>>(W1r, wt);
  gemm128_bf16<<<gemm_blocks, blk, 0, stream>>>(x, wt, xr, N);

  // ---------- layer 1: attention ----------
  fillf<<<(N * 4 + TB - 1) / TB, blk, 0, stream>>>(mx, -INFINITY, N * 4);
  fillf<<<(N * 4 + TB - 1) / TB, blk, 0, stream>>>(denom, 0.0f, N * 4);
  fillf<<<(N * CH + TB - 1) / TB, blk, 0, stream>>>(hbuf, 0.0f, N * CH);

  edge_score_h4<<<(E * CH + TB - 1) / TB, blk, 0, stream>>>(xl, xr, src, dst, att1, esc, mx, E);
  edge_exp<<<(E * 4 + TB - 1) / TB, blk, 0, stream>>>(esc, mx, denom, dst, E, 2);
  edge_agg<<<(E * CH + TB - 1) / TB, blk, 0, stream>>>(xl, esc, denom, src, dst, hbuf, E, 2, 5);
  elu_bias<<<(N * CH + TB - 1) / TB, blk, 0, stream>>>(hbuf, b1, N * CH);

  // ---------- layer 2: GEMMs (reuse xl/xr buffers as hl/hr) ----------
  transpose128<<<tw_blocks, blk, 0, stream>>>(W2l, wt);
  gemm128_bf16<<<gemm_blocks, blk, 0, stream>>>(hbuf, wt, xl, N);
  transpose128<<<tw_blocks, blk, 0, stream>>>(W2r, wt);
  gemm128_bf16<<<gemm_blocks, blk, 0, stream>>>(hbuf, wt, xr, N);

  // ---------- layer 2: attention (heads = 1) ----------
  fillf<<<(N + TB - 1) / TB, blk, 0, stream>>>(mx, -INFINITY, N);
  fillf<<<(N + TB - 1) / TB, blk, 0, stream>>>(denom, 0.0f, N);
  fillf<<<(N * CH + TB - 1) / TB, blk, 0, stream>>>(out, 0.0f, N * CH);

  edge_score_h1<<<(E * 32 + TB - 1) / TB, blk, 0, stream>>>(xl, xr, src, dst, att2, esc, mx, E);
  edge_exp<<<(E + TB - 1) / TB, blk, 0, stream>>>(esc, mx, denom, dst, E, 0);
  edge_agg<<<(E * CH + TB - 1) / TB, blk, 0, stream>>>(xl, esc, denom, src, dst, out, E, 0, 7);
  bias_add<<<(N * CH + TB - 1) / TB, blk, 0, stream>>>(out, b2, N * CH);
}
